// CosineSimilarityAttention_59708635349315
// MI455X (gfx1250) — matrix operand order repaired and checked by reading, not compiled, not run on hardware
//
#include <hip/hip_runtime.h>

// ---------------------------------------------------------------------------
// Cosine-similarity attention for MI455X (gfx1250, wave32, WMMA).
//   q,k,v : [B,H,S,D] fp32,  B=2 H=16 S=2048 D=64
//   out   : [B,H,S,D] fp32   (first 4,194,304 floats of d_out)
//   score : [B,H,S,S] fp32   (next 134,217,728 floats of d_out)
// Strategy: one workgroup (8 wave32) per 32-row q tile. Raw scores for the
// whole 32x2048 strip live in LDS (CDNA5: 320KB/WG), softmax in-place, then
// normalized scores streamed to HBM (NT stores) and PV done with WMMA.
// ---------------------------------------------------------------------------

typedef __attribute__((ext_vector_type(16))) _Float16 v16h;
typedef __attribute__((ext_vector_type(8)))  float    v8f;
typedef __attribute__((ext_vector_type(4)))  float    f32x4;   // native vec for NT stores

constexpr int  kB   = 2;
constexpr int  kH   = 16;
constexpr int  kS   = 2048;
constexpr int  kD   = 64;
constexpr int  BQ   = 32;           // q rows per workgroup
constexpr int  LSTR = kS + 4;       // padded LDS row stride (floats) - bank spread
constexpr float kTinv = 10.0f;      // 1 / TEMPERATURE
constexpr float kEps  = 1e-12f;

static __device__ __forceinline__ v8f wmma_f16(v16h a, v16h b, v8f c) {
  // D = A(16x32 f16) * B(32x16 f16) + C(16x16 f32)
  return __builtin_amdgcn_wmma_f32_16x16x32_f16(
      /*neg_a=*/false, a, /*neg_b=*/false, b,
      /*c_mod=*/(short)0, c, /*reuse_a=*/false, /*reuse_b=*/false);
}

__launch_bounds__(256, 1)
__global__ void cosattn_kernel(const float* __restrict__ q,
                               const float* __restrict__ k,
                               const float* __restrict__ v,
                               float* __restrict__ out,
                               float* __restrict__ score)
{
  extern __shared__ char smem_raw[];
  float*    sS   = (float*)smem_raw;                                   // [BQ][LSTR] raw scores -> P
  _Float16* sQ   = (_Float16*)(smem_raw + (size_t)BQ * LSTR * 4);      // [BQ][kD] normalized q (f16, *1/T)
  float*    sMax = (float*)(smem_raw + (size_t)BQ * LSTR * 4 + BQ * kD * 2);
  float*    sSum = sMax + BQ;

  const int tid  = threadIdx.x;
  const int lane = tid & 31;
  const int wid  = tid >> 5;          // 8 waves
  const int m16  = lane & 15;
  const int hi   = lane >> 4;         // half-wave select (CDNA5 frag layout)

  const int  qt = blockIdx.x;
  const long bh = (long)blockIdx.z * kH + blockIdx.y;
  const int  q0 = qt * BQ;

  const float* qp = q + bh * kS * kD;
  const float* kp = k + bh * kS * kD;
  const float* vp = v + bh * kS * kD;

  // ------- Phase 1: L2-normalize Q tile (fold in 1/T), store f16 to LDS ----
  {
    const int r = tid >> 3;           // 0..31 rows
    const int j = tid & 7;            // 8 threads per row, 8 floats each
    const float4* rowp = (const float4*)(qp + (size_t)(q0 + r) * kD + j * 8);
    float4 x0 = rowp[0], x1 = rowp[1];
    float vals[8] = {x0.x, x0.y, x0.z, x0.w, x1.x, x1.y, x1.z, x1.w};
    float ss = 0.f;
    #pragma unroll
    for (int i = 0; i < 8; ++i) ss += vals[i] * vals[i];
    ss += __shfl_xor(ss, 1, 32);
    ss += __shfl_xor(ss, 2, 32);
    ss += __shfl_xor(ss, 4, 32);
    const float scale = kTinv / fmaxf(sqrtf(ss), kEps);
    _Float16* dst = sQ + r * kD + j * 8;
    #pragma unroll
    for (int i = 0; i < 8; ++i) dst[i] = (_Float16)(vals[i] * scale);
  }
  __syncthreads();

  // ------- Phase 2: S = (q_n/T) . k_n  via WMMA, into LDS strip ------------
  // A fragment, this code's k order (not ISA table order; B matches): lane m16 = row M; halves hold
//   K = {hi*16..hi*16+15} contiguous.
  v16h aq[2][2];
  #pragma unroll
  for (int rb = 0; rb < 2; ++rb)
    #pragma unroll
    for (int ks = 0; ks < 2; ++ks) {
      const _Float16* rowq = sQ + (rb * 16 + m16) * kD + ks * 32;
      #pragma unroll
      for (int i = 0; i < 8; ++i) {
        aq[rb][ks][i]     = rowq[hi * 16 + i];
        aq[rb][ks][8 + i] = rowq[hi * 16 + 8 + i];
      }
    }

  for (int ct = wid; ct < kS / 16; ct += 8) {           // 16 iters/wave, uniform
    const int n0 = ct * 16;
    // B fragment (same k order as A): lane m16 = col N (= k row); lane group
    // selects K(=d) block of 16, contiguous within the lane.
    const float* krow = kp + (size_t)(n0 + m16) * kD;
    const float4* k4p = (const float4*)krow;
    const int base = hi * 4;                            // d0/4, d0 = hi*16
    float kr[32];
    #pragma unroll
    for (int i = 0; i < 4; ++i) {
      float4 t = k4p[base + i];                         // d = hi*16 + 4i ..
      kr[4*i+0] = t.x; kr[4*i+1] = t.y; kr[4*i+2] = t.z; kr[4*i+3] = t.w;
    }
    #pragma unroll
    for (int i = 0; i < 4; ++i) {
      float4 t = k4p[8 + base + i];                     // d = 32 + hi*16 + 4i ..
      kr[16+4*i+0] = t.x; kr[16+4*i+1] = t.y; kr[16+4*i+2] = t.z; kr[16+4*i+3] = t.w;
    }
    float ss = 0.f;
    #pragma unroll
    for (int i = 0; i < 32; ++i) ss += kr[i] * kr[i];
    ss += __shfl_xor(ss, 16, 32);                       // partner lane holds other half of row
    const float kscale = 1.0f / fmaxf(sqrtf(ss), kEps);
    v16h b0, b1;
    #pragma unroll
    for (int i = 0; i < 16; ++i) {
      b0[i] = (_Float16)(kr[i]      * kscale);          // d-block 0..31
      b1[i] = (_Float16)(kr[16 + i] * kscale);          // d-block 32..63
    }
    v8f c0 = {}; v8f c1 = {};
    c0 = wmma_f16(aq[0][0], b0, c0);
    c0 = wmma_f16(aq[0][1], b1, c0);
    c1 = wmma_f16(aq[1][0], b0, c1);
    c1 = wmma_f16(aq[1][1], b1, c1);
    // C/D layout: VGPR i -> row i + hi*8, col = m16
    #pragma unroll
    for (int i = 0; i < 8; ++i) {
      sS[(i + hi * 8)      * LSTR + n0 + m16] = c0[i];
      sS[(16 + i + hi * 8) * LSTR + n0 + m16] = c1[i];
    }
  }
  __syncthreads();

  // ------- Phase 3: row max ------------------------------------------------
  {
    const int r = tid >> 3, j = tid & 7;
    const float* rowS = sS + r * LSTR;
    float m = -3.0e38f;
    for (int c = j * 256; c < j * 256 + 256; ++c) m = fmaxf(m, rowS[c]);
    m = fmaxf(m, __shfl_xor(m, 1, 32));
    m = fmaxf(m, __shfl_xor(m, 2, 32));
    m = fmaxf(m, __shfl_xor(m, 4, 32));
    if (j == 0) sMax[r] = m;
  }
  __syncthreads();

  // ------- Phase 4: exp + row sum (in place) -------------------------------
  {
    const int r = tid >> 3, j = tid & 7;
    const float m = sMax[r];
    float* rowS = sS + r * LSTR;
    float sum = 0.f;
    for (int c = j * 256; c < j * 256 + 256; ++c) {
      float p = __expf(rowS[c] - m);
      rowS[c] = p;
      sum += p;
    }
    sum += __shfl_xor(sum, 1, 32);
    sum += __shfl_xor(sum, 2, 32);
    sum += __shfl_xor(sum, 4, 32);
    if (j == 0) sSum[r] = sum;
  }
  __syncthreads();

  // ------- Phase 5: normalize in LDS + stream score rows to HBM (NT) -------
  for (int r = 0; r < BQ; ++r) {
    const float inv = 1.0f / sSum[r];
    f32x4* rowS4 = (f32x4*)(sS + r * LSTR);                       // 16B aligned (LSTR%4==0)
    f32x4* dst4  = (f32x4*)(score + (bh * kS + q0 + r) * (long)kS);
    for (int c4 = tid; c4 < kS / 4; c4 += 256) {
      f32x4 p = rowS4[c4];
      p *= inv;
      rowS4[c4] = p;                                 // keep normalized P for PV phase
      __builtin_nontemporal_store(p, dst4 + c4);     // 537MB stream: bypass cache
    }
  }
  __syncthreads();

  // ------- Phase 6: out = P . V via WMMA -----------------------------------
  {
    const int rb = wid >> 2;                 // row block 0..1
    const int db = wid & 3;                  // d block 0..3
    const int prow = rb * 16 + m16;          // A lane row
    const int dcol = db * 16 + m16;          // B lane col
    v8f acc = {};
    for (int nk = 0; nk < kS / 32; ++nk) {   // 64 uniform iterations
      const int nb = nk * 32;
      v16h a, bv;
      const float* pr = sS + prow * LSTR + nb;
      #pragma unroll
      for (int i = 0; i < 8; ++i) {          // A frag: K={hi*16..hi*16+15}
        a[i]     = (_Float16)pr[hi * 16 + i];
        a[8 + i] = (_Float16)pr[hi * 16 + 8 + i];
      }
      const float* vcol = vp + (size_t)(nb + hi * 16) * kD + dcol;
      #pragma unroll
      for (int i = 0; i < 16; ++i)           // B frag: K contiguous per lane grp
        bv[i] = (_Float16)vcol[i * kD];
      acc = wmma_f16(a, bv, acc);
    }
    float* op = out + (bh * kS + q0) * (long)kD;
    #pragma unroll
    for (int i = 0; i < 8; ++i)
      op[(size_t)(rb * 16 + i + hi * 8) * kD + dcol] = acc[i];
  }
}

extern "C" void kernel_launch(void* const* d_in, const int* in_sizes, int n_in,
                              void* d_out, int out_size, void* d_ws, size_t ws_size,
                              hipStream_t stream) {
  (void)in_sizes; (void)n_in; (void)d_ws; (void)ws_size; (void)out_size;
  const float* q = (const float*)d_in[0];
  const float* k = (const float*)d_in[1];
  const float* v = (const float*)d_in[2];
  float* out   = (float*)d_out;
  float* score = out + (size_t)kB * kH * kS * kD;   // outputs concatenated

  const size_t shmem = (size_t)BQ * LSTR * sizeof(float)      // score strip
                     + (size_t)BQ * kD * sizeof(_Float16)     // q_norm f16
                     + (size_t)2 * BQ * sizeof(float);        // rowmax/rowsum
  (void)hipFuncSetAttribute((const void*)cosattn_kernel,
                            hipFuncAttributeMaxDynamicSharedMemorySize,
                            (int)shmem);
  dim3 grid(kS / BQ, kH, kB);   // 64 x 16 x 2 workgroups
  dim3 block(256);              // 8 wave32
  cosattn_kernel<<<grid, block, shmem, stream>>>(q, k, v, out, score);
}
